// VectorQuantizer_72722386256093
// MI455X (gfx1250) — matrix operand order repaired and checked by reading, not compiled, not run on hardware
//
#include <hip/hip_runtime.h>
#include <hip/hip_bf16.h>
#include <math.h>

// ---------------------------------------------------------------------------
// VQ-VAE vector quantizer for MI455X (gfx1250, wave32, WMMA).
//
// argmin_k ||z_n - e_k||^2 == argmax_k ( z_n . e_k - 0.5*||e_k||^2 )
// -> (M=32768) x (N=8192) x (K=256) GEMM + fused row argmax: 137 GFLOP,
//    42 MB working set (fits 192MB L2) => compute-bound => WMMA pipe.
// Precision: 2-term bf16 split of both operands, 3 WMMAs per K-step:
//    z.e ~= zh.eh + zh.el + zl.eh   (fp32 accumulate inside WMMA)
// restoring ~16 mantissa bits (top-1/2 score gaps here are ~1e-6; plain
// bf16 error ~6e-6 would scramble argmins).
//
// Round-2 fix: rounds 0/1 spilled the z-lo fragments to scratch because the
// compiler capped VGPRs for the default flat-workgroup-size (1,1024).
// __launch_bounds__(256, 1) lifts the cap so all 16 A fragments (128 VGPRs)
// + accumulator + argmax state (~190 total) stay in registers; the hot loop
// should now be pure global_load_b128 (codebook stream) + v_wmma.
// ---------------------------------------------------------------------------

typedef __bf16 bf16_t;
typedef bf16_t v16bf __attribute__((ext_vector_type(16)));
typedef bf16_t v8bfv __attribute__((ext_vector_type(8)));
typedef float  v8f   __attribute__((ext_vector_type(8)));

#define VQ_B   32
#define VQ_C   256
#define VQ_N   1024          // H*W
#define VQ_K   8192
#define VQ_BN  (VQ_B * VQ_N)         // 32768 query rows
#define VQ_TOT (VQ_BN * VQ_C)        // 8388608 elements

// ---------------------------------------------------------------------------
// Kernel 1: codebook -> bf16 hi/lo, bias[k] = -0.5*||e_k||^2 (fp32)
// ---------------------------------------------------------------------------
__global__ void vq_prep_codebook(const float* __restrict__ cb,
                                 bf16_t* __restrict__ eh,
                                 bf16_t* __restrict__ el,
                                 float*  __restrict__ bias) {
    __shared__ float red[256];
    const int k = blockIdx.x;
    const int c = threadIdx.x;
    const int o = k * VQ_C + c;
    float v  = cb[o];
    bf16_t h = (bf16_t)v;
    float  hf = (float)h;
    bf16_t l = (bf16_t)(v - hf);
    eh[o] = h;
    el[o] = l;
    red[c] = v * v;
    __syncthreads();
    for (int s = 128; s > 0; s >>= 1) {
        if (c < s) red[c] += red[c + s];
        __syncthreads();
    }
    if (c == 0) bias[k] = -0.5f * red[0];
}

// ---------------------------------------------------------------------------
// Kernel 2: z (B,C,H,W) -> z_flat (B*N, C) bf16 hi/lo (one-time transpose)
// ---------------------------------------------------------------------------
__global__ void vq_prep_z(const float* __restrict__ z,
                          bf16_t* __restrict__ zh,
                          bf16_t* __restrict__ zl) {
    const int o  = blockIdx.x * 256 + threadIdx.x;   // index into (B*N, C)
    const int c  = o & (VQ_C - 1);
    const int bn = o >> 8;
    const int n  = bn & (VQ_N - 1);
    const int b  = bn >> 10;
    float v  = z[(b * VQ_C + c) * VQ_N + n];
    bf16_t h = (bf16_t)v;
    float  hf = (float)h;
    zh[o] = h;
    zl[o] = (bf16_t)(v - hf);
}

// ---------------------------------------------------------------------------
// Kernel 3: WMMA argmax. 8 waves/block, one 16-row M-tile per wave, all 16
// A fragments (hi+lo) persistent in registers, codebook streamed from L2.
// Fragment layouts (A and B share a contiguous K order, not the ISA table's):
//   A (16-bit 16x32): lane L: row M=L%16, K chunks [kb,kb+8) & [kb+8,kb+16),
//                     kb=(L/16)*16 -> two 16B loads.
//   B (16-bit 32x16): lane L: col N=L%16, contiguous 16 K at (L/16)*16
//                     -> one 32B load.
//   C/D 16x16 f32: VGPR r: lanes 0-15 -> (M=r, N=lane),
//                          lanes 16-31 -> (M=8+r, N=lane-16).
// No divergence before the WMMAs -> EXEC all ones as required.
// ---------------------------------------------------------------------------
__global__ void __launch_bounds__(256, 1)
vq_argmax(const bf16_t* __restrict__ zh,
          const bf16_t* __restrict__ zl,
          const bf16_t* __restrict__ eh,
          const bf16_t* __restrict__ el,
          const float*  __restrict__ bias,
          int*   __restrict__ bestIdx,
          float* __restrict__ idxOutF) {
    const int wave  = threadIdx.x >> 5;
    const int lane  = threadIdx.x & 31;
    const int mtile = blockIdx.x * 8 + wave;
    const int m0    = mtile * 16;
    const int row   = m0 + (lane & 15);
    const int kb    = (lane >> 4) << 4;        // 0 or 16  (A-frag K base)
    const int bsel  = (lane >> 4) << 4;        // 0 or 16  (B-frag K chunk)

    // --- persistent A fragments (hi + lo, 128 VGPRs), each reused 512x ---
    const bf16_t* pahB = zh + row * VQ_C + kb;
    const bf16_t* palB = zl + row * VQ_C + kb;
    v16bf ah[8], al[8];
#pragma unroll
    for (int ks = 0; ks < 8; ++ks) {
        v8bfv h0 = *reinterpret_cast<const v8bfv*>(pahB + ks * 32);
        v8bfv h1 = *reinterpret_cast<const v8bfv*>(pahB + ks * 32 + 8);
        v8bfv l0 = *reinterpret_cast<const v8bfv*>(palB + ks * 32);
        v8bfv l1 = *reinterpret_cast<const v8bfv*>(palB + ks * 32 + 8);
        ah[ks] = __builtin_shufflevector(h0, h1, 0,1,2,3,4,5,6,7,8,9,10,11,12,13,14,15);
        al[ks] = __builtin_shufflevector(l0, l1, 0,1,2,3,4,5,6,7,8,9,10,11,12,13,14,15);
    }

    float bestS[8];
    int   bestI[8];
#pragma unroll
    for (int r = 0; r < 8; ++r) { bestS[r] = -3.0e38f; bestI[r] = 0; }

    // --- stream the codebook: 512 tiles of 16 codes ---
    for (int n0 = 0; n0 < VQ_K; n0 += 16) {
        const int col = n0 + (lane & 15);
        const float bv = bias[col];
        v8f acc;
#pragma unroll
        for (int i = 0; i < 8; ++i) acc[i] = bv;   // fold -0.5*||e||^2 in

        const bf16_t* pbh = eh + col * VQ_C + bsel;
        const bf16_t* pbl = el + col * VQ_C + bsel;
#pragma unroll
        for (int ks = 0; ks < 8; ++ks) {
            v16bf bh = *reinterpret_cast<const v16bf*>(pbh + ks * 32);
            v16bf bl = *reinterpret_cast<const v16bf*>(pbl + ks * 32);
            acc = __builtin_amdgcn_wmma_f32_16x16x32_bf16(
                      false, ah[ks], false, bh, (short)0, acc, false, false);
            acc = __builtin_amdgcn_wmma_f32_16x16x32_bf16(
                      false, ah[ks], false, bl, (short)0, acc, false, false);
            acc = __builtin_amdgcn_wmma_f32_16x16x32_bf16(
                      false, al[ks], false, bh, (short)0, acc, false, false);
        }
        // running per-lane argmax (strict > + ascending n == argmin tie->low k)
#pragma unroll
        for (int r = 0; r < 8; ++r) {
            if (acc[r] > bestS[r]) { bestS[r] = acc[r]; bestI[r] = col; }
        }
    }

    // --- reduce over the 16 lanes of each half-wave (N dimension) ---
#pragma unroll
    for (int off = 8; off >= 1; off >>= 1) {
#pragma unroll
        for (int r = 0; r < 8; ++r) {
            float s = __shfl_xor(bestS[r], off, 32);
            int   i = __shfl_xor(bestI[r], off, 32);
            if (s > bestS[r] || (s == bestS[r] && i < bestI[r])) {
                bestS[r] = s; bestI[r] = i;
            }
        }
    }
    if ((lane & 15) == 0) {
        const int half = lane >> 4;          // 0: rows r, 1: rows 8+r
#pragma unroll
        for (int r = 0; r < 8; ++r) {
            const int rr = m0 + half * 8 + r;
            bestIdx[rr] = bestI[r];
            idxOutF[rr] = (float)bestI[r];
        }
    }
}

// ---------------------------------------------------------------------------
// Kernel 4: z_q gather (exact fp32) into (B,C,H,W) + commitment-loss partials
// ---------------------------------------------------------------------------
__global__ void vq_gather_loss(const float* __restrict__ z,
                               const float* __restrict__ cb,
                               const int*   __restrict__ bestIdx,
                               float* __restrict__ zq_out,
                               float* __restrict__ lossPart) {
    __shared__ float red[256];
    const int o  = blockIdx.x * 256 + threadIdx.x;   // (B,C,H*W) layout
    const int n  = o & (VQ_N - 1);
    const int bc = o >> 10;
    const int c  = bc & (VQ_C - 1);
    const int b  = bc >> 8;
    const int k  = bestIdx[b * VQ_N + n];
    const float q = cb[k * VQ_C + c];
    zq_out[o] = q;
    const float d = q - z[o];
    red[threadIdx.x] = d * d;
    __syncthreads();
    for (int s = 128; s > 0; s >>= 1) {
        if (threadIdx.x < s) red[threadIdx.x] += red[threadIdx.x + s];
        __syncthreads();
    }
    if (threadIdx.x == 0) lossPart[blockIdx.x] = red[0];
}

// ---------------------------------------------------------------------------
// Kernel 5: perplexity partials via per-element multiplicity:
// each (b,n) contributes -(1/B)*log(c/B + eps) == -sum p*log(p+eps) in total
// ---------------------------------------------------------------------------
__global__ void vq_perplexity(const int* __restrict__ bestIdx,
                              float* __restrict__ perpPart) {
    __shared__ float red[256];
    const int t  = blockIdx.x * 256 + threadIdx.x;   // t = b*N + n
    const int n  = t & (VQ_N - 1);
    const int my = bestIdx[t];
    int cnt = 0;
    for (int b2 = 0; b2 < VQ_B; ++b2)
        cnt += (bestIdx[b2 * VQ_N + n] == my) ? 1 : 0;
    const float p = (float)cnt * (1.0f / (float)VQ_B);
    red[threadIdx.x] = -(1.0f / (float)VQ_B) * logf(p + 1e-10f);
    __syncthreads();
    for (int s = 128; s > 0; s >>= 1) {
        if (threadIdx.x < s) red[threadIdx.x] += red[threadIdx.x + s];
        __syncthreads();
    }
    if (threadIdx.x == 0) perpPart[blockIdx.x] = red[0];
}

// ---------------------------------------------------------------------------
// Kernel 6: deterministic final reductions -> loss mean, exp(perplexity sum)
// ---------------------------------------------------------------------------
__global__ void vq_finalize(const float* __restrict__ lossPart, int nLoss,
                            const float* __restrict__ perpPart, int nPerp,
                            float* __restrict__ lossOut,
                            float* __restrict__ perpOut) {
    __shared__ float red[256];
    float s = 0.0f;
    for (int i = threadIdx.x; i < nLoss; i += 256) s += lossPart[i];
    red[threadIdx.x] = s;
    __syncthreads();
    for (int st = 128; st > 0; st >>= 1) {
        if (threadIdx.x < st) red[threadIdx.x] += red[threadIdx.x + st];
        __syncthreads();
    }
    if (threadIdx.x == 0) lossOut[0] = red[0] / (float)VQ_TOT;
    __syncthreads();
    s = 0.0f;
    for (int i = threadIdx.x; i < nPerp; i += 256) s += perpPart[i];
    red[threadIdx.x] = s;
    __syncthreads();
    for (int st = 128; st > 0; st >>= 1) {
        if (threadIdx.x < st) red[threadIdx.x] += red[threadIdx.x + st];
        __syncthreads();
    }
    if (threadIdx.x == 0) perpOut[0] = expf(red[0]);
}

// ---------------------------------------------------------------------------
extern "C" void kernel_launch(void* const* d_in, const int* in_sizes, int n_in,
                              void* d_out, int out_size, void* d_ws, size_t ws_size,
                              hipStream_t stream) {
    const float* z  = (const float*)d_in[0];   // (32,256,32,32)
    const float* cb = (const float*)d_in[1];   // (8192,256)

    // workspace carve-up (all offsets 256B-aligned), ~42.3 MB total
    char* ws = (char*)d_ws;
    bf16_t* eh   = (bf16_t*)(ws);                         // 4,194,304 B
    bf16_t* el   = (bf16_t*)(ws + 4194304);               // 4,194,304 B
    bf16_t* zh   = (bf16_t*)(ws + 8388608);               // 16,777,216 B
    bf16_t* zl   = (bf16_t*)(ws + 25165824);              // 16,777,216 B
    float*  bias = (float*) (ws + 41943040);              // 32,768 B
    int*    bIdx = (int*)   (ws + 41975808);              // 131,072 B
    float*  lossP= (float*) (ws + 42106880);              // 131,072 B
    float*  perpP= (float*) (ws + 42237952);              // 512 B

    // d_out: [z_q (B,C,H,W) | indices (B,H,W) as float | loss | perplexity]
    float* zq_out   = (float*)d_out;
    float* idx_out  = zq_out + VQ_TOT;
    float* loss_out = idx_out + VQ_BN;
    float* perp_out = loss_out + 1;

    vq_prep_codebook<<<VQ_K, 256, 0, stream>>>(cb, eh, el, bias);
    vq_prep_z<<<VQ_TOT / 256, 256, 0, stream>>>(z, zh, zl);
    vq_argmax<<<(VQ_BN / 16) / 8, 256, 0, stream>>>(zh, zl, eh, el, bias,
                                                    bIdx, idx_out);
    vq_gather_loss<<<VQ_TOT / 256, 256, 0, stream>>>(z, cb, bIdx, zq_out, lossP);
    vq_perplexity<<<VQ_BN / 256, 256, 0, stream>>>(bIdx, perpP);
    vq_finalize<<<1, 256, 0, stream>>>(lossP, VQ_TOT / 256, perpP, VQ_BN / 256,
                                       loss_out, perp_out);
}
